// MSDeformableAttention_49744311222352
// MI455X (gfx1250) — matrix operand order repaired and checked by reading, not compiled, not run on hardware
//
#include <hip/hip_runtime.h>
#include <hip/hip_bf16.h>

// ---------------------------------------------------------------------------
// MS Deformable Attention for MI455X (gfx1250, wave32).
//   B=1, Lq=L_in=21760, C=256, NH=8, Cv=32, NL=4, NP=4,
//   shapes {128,64,32,16}^2, level starts {0,16384,20480,21504}.
//
// Pipeline:
//   split  : f32 -> (hi, lo) bf16 planes for activations (once per matrix)
//   pack   : weights -> WMMA B-fragment order, (hi, lo) bf16 planes
//   gemm   : pure-load + v_wmma_f32_16x16x32_bf16 inner loop (3-term split:
//            ah*bh + al*bh + ah*bl; dropped al*bl ~ 2^-16 rel. error)
//   sample : softmax + bilinear gather, 1 wave per (q,h), lane = channel
// ---------------------------------------------------------------------------

#define LQ   21760      // = 170 * 128
#define CDIM 256

typedef __attribute__((ext_vector_type(16))) __bf16 bf16x16;
typedef __attribute__((ext_vector_type(8)))  __bf16 bf16x8;
typedef __attribute__((ext_vector_type(4)))  __bf16 bf16x4;
typedef __attribute__((ext_vector_type(8)))  float  f32x8;

// ---------------------------------------------------------------------------
// Split f32 -> hi/lo bf16 planes. n is a multiple of 4.
// ---------------------------------------------------------------------------
__global__ __launch_bounds__(256)
void split_f32_bf16(const float* __restrict__ x,
                    __bf16* __restrict__ hi,
                    __bf16* __restrict__ lo,
                    int n)
{
    const int i = (blockIdx.x * 256 + threadIdx.x) * 4;
    if (i >= n) return;
    __builtin_prefetch(x + i + 4096, 0, 3);        // -> global_prefetch_b8

    const float4 v = *(const float4*)(x + i);
    const __bf16 h0 = (__bf16)v.x, h1 = (__bf16)v.y;
    const __bf16 h2 = (__bf16)v.z, h3 = (__bf16)v.w;
    bf16x4 H = {h0, h1, h2, h3};
    bf16x4 L = {(__bf16)(v.x - (float)h0), (__bf16)(v.y - (float)h1),
                (__bf16)(v.z - (float)h2), (__bf16)(v.w - (float)h3)};
    *(bf16x4*)(hi + i) = H;
    *(bf16x4*)(lo + i) = L;
}

// ---------------------------------------------------------------------------
// Repack W[256, N] into WMMA B-fragment order, split hi/lo:
//   packed[((nt*8 + s)*32 + lane)*16 + e] = W[(s*32 + (lane&16?16:0) + e)*N
//                                             + nt*16 + (lane&15)]
// so a wave's whole 32x16 B fragment is 32 contiguous bytes per lane.
// ---------------------------------------------------------------------------
__global__ __launch_bounds__(256)
void pack_weights(const float* __restrict__ W,
                  __bf16* __restrict__ ph,
                  __bf16* __restrict__ pl,
                  int N)               // total elems = N * 256
{
    const int idx = blockIdx.x * 256 + threadIdx.x;
    if (idx >= N * 256) return;
    const int e  = idx & 15;
    const int l  = (idx >> 4) & 31;
    const int s  = (idx >> 9) & 7;
    const int nt = idx >> 12;
    const int col = nt * 16 + (l & 15);
    const int k   = s * 32 + ((l >> 4) & 1) * 16 + e;
    const float f = W[(size_t)k * N + col];
    const __bf16 h = (__bf16)f;
    ph[idx] = h;
    pl[idx] = (__bf16)(f - (float)h);
}

// ---------------------------------------------------------------------------
// C[M,N] = A[M,256] * W[256,N] + bias[N]   (HAS_MASK: zero padded rows)
// A given as pre-split bf16 planes; W given packed+split (fragment order).
// Block: 256 thr = 8 waves; wave = one 16x16 tile; block tile = 128 x 16.
// Inner loop: 4 coalesced b128 global loads (A) + 2 b128-pair LDS reads (B)
// + 3 WMMAs per k-step, fully unrolled over K=256.
// ---------------------------------------------------------------------------
template<bool HAS_MASK>
__global__ __launch_bounds__(256)
void gemm_wmma_split(const __bf16* __restrict__ Ah,
                     const __bf16* __restrict__ Al,
                     const __bf16* __restrict__ Wph,   // packed hi
                     const __bf16* __restrict__ Wpl,   // packed lo
                     const float*  __restrict__ bias,
                     const unsigned char* __restrict__ rowmask,
                     float* __restrict__ C,
                     int M, int N)
{
    __shared__ __align__(32) __bf16 ldsH[8 * 32 * 16];   // 8 KB
    __shared__ __align__(32) __bf16 ldsL[8 * 32 * 16];   // 8 KB

    const int tid   = threadIdx.x;
    const int lane  = tid & 31;
    const int wave  = tid >> 5;
    const int tileN = blockIdx.y * 16;
    const int tileM = (blockIdx.x * 8 + wave) * 16;

    // Stage this block's packed 16-column weight panel (hi+lo) into LDS.
    {
        const uint4* sH = (const uint4*)(Wph + (size_t)blockIdx.y * 4096);
        const uint4* sL = (const uint4*)(Wpl + (size_t)blockIdx.y * 4096);
        uint4* dH = (uint4*)ldsH;
        uint4* dL = (uint4*)ldsL;
        #pragma unroll
        for (int i = 0; i < 2; ++i) {             // 512 uint4 per plane
            dH[tid + i * 256] = sH[tid + i * 256];
            dL[tid + i * 256] = sL[tid + i * 256];
        }
    }
    __syncthreads();

    const int  half = (lane >> 4) & 1;            // lanes 16..31 = hi half
    const int  rowA = tileM + (lane & 15);
    const int  colB = lane & 15;
    const __bf16* __restrict__ arh = Ah + (size_t)rowA * CDIM;
    const __bf16* __restrict__ arl = Al + (size_t)rowA * CDIM;
    const bf16x16* __restrict__ BH = (const bf16x16*)ldsH;
    const bf16x16* __restrict__ BL = (const bf16x16*)ldsL;

    f32x8 acc = {};

    #pragma unroll
    for (int s = 0; s < 8; ++s) {
        // A fragment (contiguous K order, matches packed B): lane half picks K 16-range.
        const int kA = s * 32 + half * 16;
        const bf16x8 ah0 = *(const bf16x8*)(arh + kA);
        const bf16x8 ah1 = *(const bf16x8*)(arh + kA + 8);
        const bf16x8 al0 = *(const bf16x8*)(arl + kA);
        const bf16x8 al1 = *(const bf16x8*)(arl + kA + 8);
        const bf16x16 ah = __builtin_shufflevector(ah0, ah1,
                 0,1,2,3,4,5,6,7,8,9,10,11,12,13,14,15);
        const bf16x16 al = __builtin_shufflevector(al0, al1,
                 0,1,2,3,4,5,6,7,8,9,10,11,12,13,14,15);

        // B fragment: one 32 B contiguous LDS read per plane.
        const bf16x16 bh = BH[s * 32 + lane];
        const bf16x16 bl = BL[s * 32 + lane];

        acc = __builtin_amdgcn_wmma_f32_16x16x32_bf16(
                  false, ah, false, bh, (short)0, acc, false, false);
        acc = __builtin_amdgcn_wmma_f32_16x16x32_bf16(
                  false, al, false, bh, (short)0, acc, false, false);
        acc = __builtin_amdgcn_wmma_f32_16x16x32_bf16(
                  false, ah, false, bl, (short)0, acc, false, false);
    }

    // Epilogue: D VGPR r holds row (tileM + r + 8*half), col = tileN + colB.
    const float bval = bias[tileN + colB];
    #pragma unroll
    for (int r = 0; r < 8; ++r) {
        const int row = tileM + r + half * 8;
        float v = acc[r] + bval;
        if (HAS_MASK && rowmask[row]) v = 0.0f;
        if (row < M)
            C[(size_t)row * N + tileN + colB] = v;
    }
}

// ---------------------------------------------------------------------------
// Softmax + multi-scale bilinear sampling.
// One wave per (query, head); lane = channel (Cv = 32 == wave32) -> every
// corner fetch is one coalesced 128 B transaction (value is L2-resident).
// ---------------------------------------------------------------------------
__global__ __launch_bounds__(256)
void msda_sample_kernel(const float* __restrict__ value,   // [L, 256]
                        const float* __restrict__ refpts,  // [Lq, 4, 2]
                        const float* __restrict__ off,     // [Lq, 256]
                        const float* __restrict__ attn,    // [Lq, 128]
                        float* __restrict__ out)           // [Lq, 256]
{
    const int gwave = (blockIdx.x * 256 + threadIdx.x) >> 5;
    const int lane  = threadIdx.x & 31;
    const int q = gwave >> 3;
    const int h = gwave & 7;
    if (q >= LQ) return;

    const int ch = h * 32 + lane;

    float logits[16];
    const float* ap = attn + (size_t)q * 128 + h * 16;
    float m = -3.0e38f;
    #pragma unroll
    for (int p = 0; p < 16; ++p) { logits[p] = ap[p]; m = fmaxf(m, logits[p]); }
    float denom = 0.0f;
    #pragma unroll
    for (int p = 0; p < 16; ++p) { logits[p] = __expf(logits[p] - m); denom += logits[p]; }
    const float inv_denom = __frcp_rn(denom);

    const float* op = off + (size_t)q * 256 + h * 32;

    const int Hs[4]  = {128, 64, 32, 16};
    const int Ws[4]  = {128, 64, 32, 16};
    const int LSI[4] = {0, 16384, 20480, 21504};

    float acc = 0.0f;

    #pragma unroll
    for (int l = 0; l < 4; ++l) {
        const int   Hl = Hs[l], Wl = Ws[l], base = LSI[l];
        const float fH = (float)Hl, fW = (float)Wl;
        const float rx = refpts[(size_t)q * 8 + l * 2 + 0];
        const float ry = refpts[(size_t)q * 8 + l * 2 + 1];

        #pragma unroll
        for (int p = 0; p < 4; ++p) {
            const float ox = op[(l * 4 + p) * 2 + 0];
            const float oy = op[(l * 4 + p) * 2 + 1];
            const float gx = rx + ox / fW;
            const float gy = ry + oy / fH;

            const float x  = gx * fW - 0.5f;
            const float y  = gy * fH - 0.5f;
            const float x0 = floorf(x);
            const float y0 = floorf(y);
            const float dx = x - x0;
            const float dy = y - y0;
            const int xi = (int)x0, yi = (int)y0;

            const float w00 = (1.0f - dx) * (1.0f - dy);
            const float w01 = dx * (1.0f - dy);
            const float w10 = (1.0f - dx) * dy;
            const float w11 = dx * dy;

            float s = 0.0f;
            if (yi >= 0 && yi < Hl && xi >= 0 && xi < Wl)
                s += w00 * value[(size_t)(base + yi * Wl + xi) * CDIM + ch];
            if (yi >= 0 && yi < Hl && (xi + 1) < Wl)
                s += w01 * value[(size_t)(base + yi * Wl + xi + 1) * CDIM + ch];
            if ((yi + 1) < Hl && xi >= 0 && xi < Wl)
                s += w10 * value[(size_t)(base + (yi + 1) * Wl + xi) * CDIM + ch];
            if ((yi + 1) < Hl && (xi + 1) < Wl && (xi + 1) >= 0 && (yi + 1) >= 0)
                s += w11 * value[(size_t)(base + (yi + 1) * Wl + xi + 1) * CDIM + ch];

            acc += (logits[l * 4 + p] * inv_denom) * s;
        }
    }

    out[(size_t)q * 256 + ch] = acc;
}

// ---------------------------------------------------------------------------
extern "C" void kernel_launch(void* const* d_in, const int* in_sizes, int n_in,
                              void* d_out, int out_size, void* d_ws, size_t ws_size,
                              hipStream_t stream)
{
    (void)in_sizes; (void)n_in; (void)out_size; (void)ws_size;

    const float*         query   = (const float*)d_in[0];
    const float*         refpts  = (const float*)d_in[1];
    const float*         inflat  = (const float*)d_in[2];
    const unsigned char* padmask = (const unsigned char*)d_in[5];
    const float*         W_off   = (const float*)d_in[6];
    const float*         b_off   = (const float*)d_in[7];
    const float*         W_attn  = (const float*)d_in[8];
    const float*         b_attn  = (const float*)d_in[9];
    const float*         W_v     = (const float*)d_in[10];
    const float*         b_v     = (const float*)d_in[11];
    const float*         W_out   = (const float*)d_in[12];
    const float*         b_out   = (const float*)d_in[13];
    float*               out     = (float*)d_out;

    // ---- workspace carve-out (~146 MB, all rewritten every call) ----------
    char* p = (char*)d_ws;
    auto carve = [&](size_t bytes) {
        void* r = (void*)p;
        p += (bytes + 255) & ~(size_t)255;
        return r;
    };
    const size_t NE = (size_t)LQ * CDIM;                 // 5,570,560

    float*  value = (float*) carve(NE * 4);
    float*  off   = (float*) carve(NE * 4);
    float*  attn  = (float*) carve((size_t)LQ * 128 * 4);
    float*  mid   = (float*) carve(NE * 4);
    __bf16* Xh    = (__bf16*)carve(NE * 2);
    __bf16* Xl    = (__bf16*)carve(NE * 2);
    __bf16* Qh    = (__bf16*)carve(NE * 2);
    __bf16* Ql    = (__bf16*)carve(NE * 2);
    __bf16* Mh    = (__bf16*)carve(NE * 2);
    __bf16* Ml    = (__bf16*)carve(NE * 2);
    __bf16* Wvh   = (__bf16*)carve(65536 * 2);
    __bf16* Wvl   = (__bf16*)carve(65536 * 2);
    __bf16* Woh   = (__bf16*)carve(65536 * 2);
    __bf16* Wol   = (__bf16*)carve(65536 * 2);
    __bf16* Wah   = (__bf16*)carve(32768 * 2);
    __bf16* Wal   = (__bf16*)carve(32768 * 2);
    __bf16* Wuh   = (__bf16*)carve(65536 * 2);
    __bf16* Wul   = (__bf16*)carve(65536 * 2);

    const dim3 blk(256);
    const int  mblocks   = LQ / 128;                      // 170, exact
    const int  splitBlks = (int)(NE / 4 / 256);           // 5440, exact

    // ---- precision split + weight packing --------------------------------
    split_f32_bf16<<<dim3(splitBlks), blk, 0, stream>>>(inflat, Xh, Xl, (int)NE);
    split_f32_bf16<<<dim3(splitBlks), blk, 0, stream>>>(query,  Qh, Ql, (int)NE);
    pack_weights<<<dim3(256), blk, 0, stream>>>(W_v,   Wvh, Wvl, 256);
    pack_weights<<<dim3(256), blk, 0, stream>>>(W_off, Woh, Wol, 256);
    pack_weights<<<dim3(128), blk, 0, stream>>>(W_attn, Wah, Wal, 128);
    pack_weights<<<dim3(256), blk, 0, stream>>>(W_out, Wuh, Wul, 256);

    // ---- stage 1: projections --------------------------------------------
    gemm_wmma_split<true ><<<dim3(mblocks, 16), blk, 0, stream>>>(
        Xh, Xl, Wvh, Wvl, b_v, padmask, value, LQ, 256);
    gemm_wmma_split<false><<<dim3(mblocks, 16), blk, 0, stream>>>(
        Qh, Ql, Woh, Wol, b_off, nullptr, off, LQ, 256);
    gemm_wmma_split<false><<<dim3(mblocks, 8), blk, 0, stream>>>(
        Qh, Ql, Wah, Wal, b_attn, nullptr, attn, LQ, 128);

    // ---- stage 2: softmax + bilinear gather ------------------------------
    msda_sample_kernel<<<dim3(LQ), blk, 0, stream>>>(value, refpts, off, attn, mid);

    // ---- stage 3: output projection --------------------------------------
    split_f32_bf16<<<dim3(splitBlks), blk, 0, stream>>>(mid, Mh, Ml, (int)NE);
    gemm_wmma_split<false><<<dim3(mblocks, 16), blk, 0, stream>>>(
        Mh, Ml, Wuh, Wul, b_out, nullptr, out, LQ, 256);
}
